// GATBlock_73778948211131
// MI455X (gfx1250) — matrix operand order repaired and checked by reading, not compiled, not run on hardware
//
#include <hip/hip_runtime.h>

typedef __attribute__((ext_vector_type(16))) _Float16 v16h;
typedef __attribute__((ext_vector_type(8)))  _Float16 v8h;
typedef __attribute__((ext_vector_type(8)))  float    v8f;

#define B_    8
#define CIN   64
#define COUT  64
#define H_    128
#define W_    128
#define N_    (H_ * W_)        // 16384
#define HEADS 4
#define CR    32
#define FDIM  (HEADS * CR)     // 128

// Workspace layout (bytes)
static constexpr size_t WC_OFF = 0;                                   // 128x64 f16 combined lin@reduce
static constexpr size_t WR_OFF = 64 * 1024;                           // 64x32 f16 restore weights
static constexpr size_t ST_OFF = 128 * 1024;                          // 128 f32: sum[64], sumsq[64]
static constexpr size_t HF_OFF = 1ull << 20;                          // B*N*128 f16 (32 MB)
static constexpr size_t G_OFF  = HF_OFF + (size_t)B_ * N_ * FDIM * 2; // B*N*32 f16 (8 MB)

union FragH {
    v16h v;
    v8h  h8[2];
    _Float16 h[16];
};

// ---------------------------------------------------------------------------
// prep: Wc = w_lin @ w_reduce (f16), Wr = f16(w_restore), zero BN stats
// ---------------------------------------------------------------------------
__global__ void prep_kernel(const float* __restrict__ w_reduce,
                            const float* __restrict__ w_lin,
                            const float* __restrict__ w_restore,
                            char* __restrict__ ws) {
    _Float16* Wc = (_Float16*)(ws + WC_OFF);
    _Float16* Wr = (_Float16*)(ws + WR_OFF);
    float*    st = (float*)(ws + ST_OFF);
    int tid = threadIdx.x;
    for (int idx = tid; idx < FDIM * CIN; idx += 256) {
        int o = idx / CIN, c = idx % CIN;
        float s = 0.f;
        for (int r = 0; r < CR; ++r) s += w_lin[o * CR + r] * w_reduce[r * CIN + c];
        Wc[idx] = (_Float16)s;
    }
    for (int idx = tid; idx < COUT * CR; idx += 256) Wr[idx] = (_Float16)w_restore[idx];
    for (int idx = tid; idx < 2 * COUT; idx += 256) st[idx] = 0.f;
}

// ---------------------------------------------------------------------------
// GEMM1: hfeat[b,n,0:128] = f16(x_nodes[b,n,0:64]) @ Wc^T   (M=nodes)
// ---------------------------------------------------------------------------
__global__ __launch_bounds__(256) void gemm_hfeat(const float* __restrict__ x,
                                                  char* __restrict__ ws) {
    __shared__ __align__(16) _Float16 Xs[128 * 72];   // node-major x tile (stride 72 halves)
    __shared__ __align__(16) _Float16 Ws[FDIM * CIN]; // combined weights
    const _Float16* Wc    = (const _Float16*)(ws + WC_OFF);
    _Float16*       hfeat = (_Float16*)(ws + HF_OFF);

    int blk = blockIdx.x;
    int b   = blk >> 7;            // 128 blocks per batch
    int n0  = (blk & 127) << 7;    // 128 nodes per block
    const float* xb = x + (size_t)b * CIN * N_;
    int tid = threadIdx.x;

    for (int idx = tid; idx < CIN * 128; idx += 256) {
        int c = idx >> 7, n = idx & 127;              // coalesced along n
        Xs[n * 72 + c] = (_Float16)xb[(size_t)c * N_ + n0 + n];
    }
    for (int idx = tid; idx < FDIM * CIN; idx += 256) Ws[idx] = Wc[idx];
    __syncthreads();

    int wave = tid >> 5, lane = tid & 31;
    int m  = 16 * wave + (lane & 15);
    int b1 = (lane < 16) ? 0 : 8;   // A-frag K base (ISA 16-bit A layout)
    FragH a0, a1;
    {
        const _Float16* row = &Xs[m * 72];
        a0.h8[0] = *(const v8h*)(row + b1);        // K = b1 .. b1+7
        a0.h8[1] = *(const v8h*)(row + b1 + 16);   // K = b1+16 .. b1+23
        a1.h8[0] = *(const v8h*)(row + 32 + b1);
        a1.h8[1] = *(const v8h*)(row + 32 + b1 + 16);
    }
    int bk = (lane < 16) ? 0 : 8;   // B-frag K base (same layout as A: K = bk..bk+7, bk+16..bk+23)
    for (int ct = 0; ct < 8; ++ct) {
        int o = ct * 16 + (lane & 15);
        const _Float16* wrow = &Ws[o * CIN];
        FragH bf0, bf1;
        bf0.h8[0] = *(const v8h*)(wrow + bk);
        bf0.h8[1] = *(const v8h*)(wrow + bk + 16);
        bf1.h8[0] = *(const v8h*)(wrow + 32 + bk);
        bf1.h8[1] = *(const v8h*)(wrow + 32 + bk + 16);
        v8f acc = {0.f, 0.f, 0.f, 0.f, 0.f, 0.f, 0.f, 0.f};
        acc = __builtin_amdgcn_wmma_f32_16x16x32_f16(false, a0.v, false, bf0.v,
                                                     (short)0, acc, false, false);
        acc = __builtin_amdgcn_wmma_f32_16x16x32_f16(false, a1.v, false, bf1.v,
                                                     (short)0, acc, false, false);
        // C layout: row M = v + 8*(lane>=16), col = lane&15
        int col  = ct * 16 + (lane & 15);
        int mrow = 16 * wave + ((lane < 16) ? 0 : 8);
        _Float16* outp = hfeat + ((size_t)b * N_ + n0 + mrow) * FDIM + col;
#pragma unroll
        for (int v = 0; v < 8; ++v) outp[(size_t)v * FDIM] = (_Float16)acc[v];
    }
}

// ---------------------------------------------------------------------------
// Fused GAT: per-dst 5-point softmax over {self, up, down, left, right}.
// Phase 1 computes attention logits (dot products with att vectors) directly
// from the neighbor hfeat rows; phase 2 re-reads the rows (L2-resident:
// hfeat = 33.5 MB << 192 MB L2) for the weighted aggregation.
// g[n, 0:32] = mean_h sum_src alpha * hfeat[src, h*32 + c] + b_gat
// ---------------------------------------------------------------------------
__global__ __launch_bounds__(256) void gat_stencil(const float* __restrict__ att_src,
                                                   const float* __restrict__ att_dst,
                                                   const float* __restrict__ b_gat,
                                                   char* __restrict__ ws) {
    __shared__ float As[FDIM];
    __shared__ float Ad[FDIM];
    __shared__ float Bg[CR];
    {
        int tid = threadIdx.x;
        for (int idx = tid; idx < FDIM; idx += 256) {
            As[idx] = att_src[idx];
            Ad[idx] = att_dst[idx];
        }
        if (tid < CR) Bg[tid] = b_gat[tid];
    }
    __syncthreads();

    size_t t = (size_t)blockIdx.x * 256 + threadIdx.x;
    int b = (int)(t / N_);
    int n = (int)(t % N_);
    int i = n >> 7, j = n & 127;   // W_ = 128
    const _Float16* hfeat = (const _Float16*)(ws + HF_OFF);
    _Float16* g = (_Float16*)(ws + G_OFF);
    const size_t rowbase = (size_t)b * N_;

    int srcs[5];
    int ns = 0;
    srcs[ns++] = n;                 // self first (also supplies a_d)
    if (i > 0)      srcs[ns++] = n - W_;
    if (i < H_ - 1) srcs[ns++] = n + W_;
    if (j > 0)      srcs[ns++] = n - 1;
    if (j < W_ - 1) srcs[ns++] = n + 1;

    // Phase 1: attention logits from neighbor rows
    float as_k[5][HEADS];
    float ad_d[HEADS];
    for (int k = 0; k < ns; ++k) {
        const _Float16* hr = hfeat + (rowbase + srcs[k]) * FDIM;
#pragma unroll
        for (int h = 0; h < HEADS; ++h) {
            float ss = 0.f, sd = 0.f;
#pragma unroll
            for (int c = 0; c < CR; ++c) {
                float v = (float)hr[h * CR + c];
                ss += v * As[h * CR + c];
                sd += v * Ad[h * CR + c];
            }
            as_k[k][h] = ss;
            if (k == 0) ad_d[h] = sd;   // dst == self row
        }
    }

    // Softmax over incoming edges, per head
    float alpha[HEADS][5];
#pragma unroll
    for (int h = 0; h < HEADS; ++h) {
        float e[5], m = -1e30f;
        for (int k = 0; k < ns; ++k) {
            float v = as_k[k][h] + ad_d[h];
            v = (v > 0.f) ? v : 0.2f * v;   // leaky_relu(0.2)
            e[k] = v;
            m = fmaxf(m, v);
        }
        float den = 0.f;
        for (int k = 0; k < ns; ++k) { e[k] = __expf(e[k] - m); den += e[k]; }
        float inv = 1.f / den;
        for (int k = 0; k < ns; ++k) alpha[h][k] = e[k] * inv;
    }

    // Phase 2: weighted aggregation (rows hit in L0/L2)
    float acc[CR];
#pragma unroll
    for (int c = 0; c < CR; ++c) acc[c] = 0.f;
    for (int k = 0; k < ns; ++k) {
        const _Float16* hr = hfeat + (rowbase + srcs[k]) * FDIM;
#pragma unroll
        for (int h = 0; h < HEADS; ++h) {
            float w = alpha[h][k];
#pragma unroll
            for (int c = 0; c < CR; ++c) acc[c] += w * (float)hr[h * CR + c];
        }
    }
    _Float16* go = g + t * CR;
#pragma unroll
    for (int c = 0; c < CR; ++c) go[c] = (_Float16)(acc[c] * 0.25f + Bg[c]);
}

// ---------------------------------------------------------------------------
// GEMM3: out[b,cout,n] = w_restore @ g^T + x   (M=cout so stores coalesce on n)
// ---------------------------------------------------------------------------
__global__ __launch_bounds__(256) void gemm_restore(const float* __restrict__ x,
                                                    float* __restrict__ out,
                                                    char* __restrict__ ws) {
    __shared__ __align__(16) _Float16 WrS[COUT * CR];
    const _Float16* Wr = (const _Float16*)(ws + WR_OFF);
    const _Float16* g  = (const _Float16*)(ws + G_OFF);

    int blk = blockIdx.x;
    int b   = blk >> 7;
    int n0  = (blk & 127) << 7;
    int tid = threadIdx.x;
    for (int idx = tid; idx < COUT * CR; idx += 256) WrS[idx] = Wr[idx];
    __syncthreads();

    int wave = tid >> 5, lane = tid & 31;
    int rt  = wave & 3;            // cout tile [16*rt, 16*rt+16)
    int ct0 = (wave >> 2) * 4;     // node tile base
    int b1  = (lane < 16) ? 0 : 8;
    FragH a;   // A = w_restore 16x32 (M=cout, K=cr)
    {
        const _Float16* row = &WrS[(rt * 16 + (lane & 15)) * CR];
        a.h8[0] = *(const v8h*)(row + b1);
        a.h8[1] = *(const v8h*)(row + b1 + 16);
    }
    int bk = (lane < 16) ? 0 : 8;
    for (int c = 0; c < 4; ++c) {
        int ct   = ct0 + c;
        int node = n0 + ct * 16 + (lane & 15);
        const _Float16* grow = g + ((size_t)b * N_ + node) * CR + bk;
        FragH bf;  // B = g^T 32x16 (K=cr, cols=nodes), K = bk..bk+7, bk+16..bk+23 per lane
        bf.h8[0] = *(const v8h*)(grow);
        bf.h8[1] = *(const v8h*)(grow + 16);
        v8f acc = {0.f, 0.f, 0.f, 0.f, 0.f, 0.f, 0.f, 0.f};
        acc = __builtin_amdgcn_wmma_f32_16x16x32_f16(false, a.v, false, bf.v,
                                                     (short)0, acc, false, false);
        int cout0 = rt * 16 + ((lane < 16) ? 0 : 8);
        size_t base = ((size_t)b * COUT + cout0) * N_ + node;
#pragma unroll
        for (int v = 0; v < 8; ++v) {
            size_t addr = base + (size_t)v * N_;
            out[addr] = acc[v] + x[addr];   // residual (CIN==COUT, same layout)
        }
    }
}

// ---------------------------------------------------------------------------
// BN statistics: one block per (batch, channel) slice; private accumulation,
// tree reduce in LDS, single global atomicAdd per block.
// ---------------------------------------------------------------------------
__global__ __launch_bounds__(256) void bn_stats(const float* __restrict__ out,
                                                char* __restrict__ ws) {
    __shared__ float ssum[256];
    __shared__ float ssq[256];
    float* st = (float*)(ws + ST_OFF);
    int tid   = threadIdx.x;
    int slice = blockIdx.x;                 // b*COUT + ch
    const float* p = out + (size_t)slice * N_;
    float s = 0.f, q = 0.f;
    for (int k = tid; k < N_; k += 256) {
        float v = p[k];
        s += v;
        q += v * v;
    }
    ssum[tid] = s;
    ssq[tid]  = q;
    __syncthreads();
    for (int off = 128; off > 0; off >>= 1) {
        if (tid < off) {
            ssum[tid] += ssum[tid + off];
            ssq[tid]  += ssq[tid + off];
        }
        __syncthreads();
    }
    if (tid == 0) {
        int ch = slice & (COUT - 1);
        atomicAdd(&st[ch], ssum[0]);
        atomicAdd(&st[COUT + ch], ssq[0]);
    }
}

// ---------------------------------------------------------------------------
// BN apply + ReLU (in place on d_out)
// ---------------------------------------------------------------------------
__global__ void bn_apply(float* __restrict__ out,
                         const float* __restrict__ gamma,
                         const float* __restrict__ beta,
                         const char* __restrict__ ws) {
    const float* st = (const float*)(ws + ST_OFF);
    size_t idx = (size_t)blockIdx.x * 256 + threadIdx.x;
    int ch = (int)((idx / N_) & (COUT - 1));
    const float cnt = (float)((size_t)B_ * N_);
    float mean = st[ch] / cnt;
    float var  = st[COUT + ch] / cnt - mean * mean;
    float v = (out[idx] - mean) * rsqrtf(var + 1e-5f) * gamma[ch] + beta[ch];
    out[idx] = fmaxf(v, 0.f);
}

// ---------------------------------------------------------------------------
extern "C" void kernel_launch(void* const* d_in, const int* in_sizes, int n_in,
                              void* d_out, int out_size, void* d_ws, size_t ws_size,
                              hipStream_t stream) {
    const float* x         = (const float*)d_in[0];
    const float* w_reduce  = (const float*)d_in[1];
    const float* w_lin     = (const float*)d_in[2];
    const float* att_src   = (const float*)d_in[3];
    const float* att_dst   = (const float*)d_in[4];
    const float* b_gat     = (const float*)d_in[5];
    const float* w_restore = (const float*)d_in[6];
    const float* bn_gamma  = (const float*)d_in[7];
    const float* bn_beta   = (const float*)d_in[8];
    float* out = (float*)d_out;
    char*  ws  = (char*)d_ws;

    const int nodes   = B_ * N_;                 // 131072
    const int gblocks = nodes / 128;             // 1024 (GEMM tiles)

    prep_kernel<<<1, 256, 0, stream>>>(w_reduce, w_lin, w_restore, ws);
    gemm_hfeat<<<gblocks, 256, 0, stream>>>(x, ws);
    gat_stencil<<<nodes / 256, 256, 0, stream>>>(att_src, att_dst, b_gat, ws);
    gemm_restore<<<gblocks, 256, 0, stream>>>(x, out, ws);
    bn_stats<<<B_ * COUT, 256, 0, stream>>>(out, ws);
    bn_apply<<<(B_ * COUT * N_) / 256, 256, 0, stream>>>(out, bn_gamma, bn_beta, ws);
}
